// NodeEquiModel_36266703848155
// MI455X (gfx1250) — matrix operand order repaired and checked by reading, not compiled, not run on hardware
//
#include <hip/hip_runtime.h>

// MI455X / gfx1250, wave32. D = A×B+C via v_wmma_f32_16x16x32_bf16.
//
// tp[ec,k] = (1/45) * Σ_{i,j} fn[ec,i]·fe[ec,j]·W_tp[i,j,k]
// implicit GEMM with K' = 64*i + j (j padded 45->64): M=2E, K=2880, N=32.
// Single i per k-step  -> fn is a per-lane scalar broadcast (1 ds_load / 2 ksteps)
// j in two aligned runs -> fe via 4x ds_load_b128; no div/mod, no cndmask padding.
// out = ((tp @ Wfc1)/sqrt(32)) @ Wfc2 / 8, fused as 10 more WMMAs per tile.
//
// One wave = one 16-row (edge-channel) tile: 180 main + 10 epilogue WMMAs.
// Workgroup = 8 waves = 128 edge-channels = 64 edges.

#define MBQ    81
#define REP    45
#define OUTK   32
#define HIDN   64
#define JPAD   64
#define KSTEPS (REP * 2)     // 90: K' = 45*64 = 2880 = 90*32

typedef __attribute__((ext_vector_type(16))) __bf16 bf16x16;
typedef __attribute__((ext_vector_type(8)))  float  f32x8;

// 81 -> 45 equi-rep gather map: voigt(s), voigt(p), voigt(d), sp, sd, pd
__constant__ int MAP45[REP] = {
    0,
    10, 20, 30, 11, 12, 21,
    40, 50, 60, 70, 80, 41, 42, 43, 44, 51, 52, 53, 61, 62, 71,
    1, 2, 3,
    4, 5, 6, 7, 8,
    13, 14, 15, 16, 17, 22, 23, 24, 25, 26, 31, 32, 33, 34, 35
};

// ---- LDS layout (bytes, dynamic shared) ----
#define WM_OFF     0
#define WM_BYTES   (KSTEPS * 2 * 512 * 2)        // 184320: 180 B-fragments, swizzled
#define WFC1_OFF   (WM_OFF + WM_BYTES)
#define WFC1_BYTES (4 * 512 * 2)                 // 4096
#define WFC2_OFF   (WFC1_OFF + WFC1_BYTES)
#define WFC2_BYTES (6 * 512 * 2)                 // 6144
#define FN_OFF     (WFC2_OFF + WFC2_BYTES)       // 194560: fn[128][45] f32
#define FN_BYTES   (128 * REP * 4)               // 23040
#define FE_OFF     (FN_OFF + FN_BYTES)           // 217600 (256B aligned): fe[128][64] f32
#define FE_BYTES   (128 * JPAD * 4)              // 32768
#define TP_OFF     FN_OFF                        // overlay after barrier (fn/fe dead)
#define H_OFF      (TP_OFF + 8 * 16 * 32 * 4)    // +16384
#define SMEM_BYTES (FE_OFF + FE_BYTES)           // 250368 (244.5KB < 320KB/WGP)

__device__ inline bf16x16 load_frag16(const __bf16* p) {
    union { bf16x16 v; float4 q[2]; } u;
    const float4* f = (const float4*)p;          // 16B-aligned ds_load_b128 pairs
    u.q[0] = f[0];
    u.q[1] = f[1];
    return u.v;
}

__global__ __launch_bounds__(256, 1)
void node_equi_tp_wmma(const float* __restrict__ f_nodes,
                       const float* __restrict__ f_edges,
                       const int*   __restrict__ edge_index,
                       const float* __restrict__ W_tp,
                       const float* __restrict__ W_fc1,
                       const float* __restrict__ W_fc2,
                       float* __restrict__ out,
                       int n_edges)
{
    extern __shared__ char smem[];
    const int tid  = threadIdx.x;
    const int lane = tid & 31;
    const int w    = tid >> 5;          // wave 0..7
    const int half = lane >> 4;         // lane half (wave32)
    const int lm   = lane & 15;
    const long necc = (long)n_edges * 2;

    // ---------- 1) W_tp -> bf16 B-fragments in K' = 64*i + j order ----------
    // B layout (32x16, 16-bit): slot s of lane half h holds K = s + (s&8) + 8*h,
    // N = lane%16. Fragment(t,nt) = 512 bf16 at frag*512.
    {
        __bf16* wm = (__bf16*)(smem + WM_OFF);
        for (int idx = tid; idx < KSTEPS * 2 * 512; idx += 256) {
            int frag = idx >> 9;                 // t*2 + nt
            int fl   = (idx >> 4) & 31;          // lane
            int fs   = idx & 15;                 // slot
            int tk   = frag >> 1;
            int nt   = frag & 1;
            int kk   = tk * 32 + fs + (fs & 8) + ((fl >> 4) << 3);   // K' row
            int i    = kk >> 6;                  // always < 45
            int j    = kk & 63;
            int nn   = nt * 16 + (fl & 15);      // k (0..31)
            float v  = (j < REP) ? W_tp[(i * REP + j) * OUTK + nn] : 0.f;
            wm[frag * 512 + fl * 16 + fs] = (__bf16)v;
        }
    }
    // ---------- 2) W_fc1 fragments (K=32 exact, 4 n-tiles) ----------
    {
        __bf16* wf = (__bf16*)(smem + WFC1_OFF);
        for (int idx = tid; idx < 4 * 512; idx += 256) {
            int frag = idx >> 9;
            int fl   = (idx >> 4) & 31;
            int fs   = idx & 15;
            int kk   = fs + (fs & 8) + ((fl >> 4) << 3);
            int nn   = frag * 16 + (fl & 15);
            wf[frag * 512 + fl * 16 + fs] = (__bf16)W_fc1[kk * HIDN + nn];
        }
    }
    // ---------- 3) W_fc2 fragments (K=64 -> 2 ksteps, N=45 pad 48 -> 3 n-tiles) ----------
    {
        __bf16* wf = (__bf16*)(smem + WFC2_OFF);
        for (int idx = tid; idx < 6 * 512; idx += 256) {
            int frag = idx >> 9;                 // kt*3 + nt
            int fl   = (idx >> 4) & 31;
            int fs   = idx & 15;
            int kt   = frag / 3, nt = frag % 3;
            int kk   = kt * 32 + fs + (fs & 8) + ((fl >> 4) << 3);
            int nn   = nt * 16 + (fl & 15);
            float v  = (nn < REP) ? W_fc2[kk * REP + nn] : 0.f;
            wf[frag * 512 + fl * 16 + fs] = (__bf16)v;
        }
    }
    // ---------- 4) gather fn (45) and fe (padded to 64) rows into LDS ----------
    {
        float* fnL = (float*)(smem + FN_OFF);
        const long ecc0 = (long)blockIdx.x * 128;
        for (int idx = tid; idx < 128 * REP; idx += 256) {
            int r = idx / REP;
            int i = idx % REP;
            long ecc = ecc0 + r;
            if (ecc >= necc) ecc = necc - 1;     // clamp (store-guarded later)
            long e = ecc >> 1;
            int  c = (int)(ecc & 1);
            int node = edge_index[e];            // row = edge_index[0][e]
            fnL[idx] = f_nodes[(long)node * (2 * MBQ) + c * MBQ + MAP45[i]];
        }
        float* feL = (float*)(smem + FE_OFF);
        for (int idx = tid; idx < 128 * JPAD; idx += 256) {
            int r = idx >> 6;
            int j = idx & 63;
            long ecc = ecc0 + r;
            if (ecc >= necc) ecc = necc - 1;
            long e = ecc >> 1;
            int  c = (int)(ecc & 1);
            feL[idx] = (j < REP) ? f_edges[e * (2 * MBQ) + c * MBQ + MAP45[j]] : 0.f;
        }
    }
    __syncthreads();

    // ---------- 5) main GEMM: 45 i-groups x 2 ksteps x 2 ntiles of WMMA ----------
    // A[m, K'] = fn[m][i] * fe_pad[m][j];  i = t>>1 (scalar per kstep),
    // j = (t&1)*32 + s + (s&8) + 8*half -> two aligned 8-float runs per lane.
    const float* fnrow = (const float*)(smem + FN_OFF) + (w * 16 + lm) * REP;
    const float* ferow = (const float*)(smem + FE_OFF) + (w * 16 + lm) * JPAD;
    const __bf16* wm   = (const __bf16*)(smem + WM_OFF);

    f32x8 acc0 = {};
    f32x8 acc1 = {};
    for (int g = 0; g < REP; ++g) {
        const float fnv = fnrow[g];              // one scalar per i-group
#pragma unroll
        for (int u = 0; u < 2; ++u) {
            const int t = g * 2 + u;
            const float* fp = ferow + u * 32 + (half << 3);
            float fv[16];
            *(float4*)&fv[0]  = *(const float4*)(fp + 0);    // j0 .. j0+3
            *(float4*)&fv[4]  = *(const float4*)(fp + 4);    // j0+4 .. j0+7
            *(float4*)&fv[8]  = *(const float4*)(fp + 16);   // j0+16 .. j0+19
            *(float4*)&fv[12] = *(const float4*)(fp + 20);   // j0+20 .. j0+23
            bf16x16 a;
#pragma unroll
            for (int s = 0; s < 16; ++s)
                a[s] = (__bf16)(fnv * fv[s]);    // v_fma_mixlo/hi_bf16
            bf16x16 b0 = load_frag16(wm + (t * 2 + 0) * 512 + lane * 16);
            bf16x16 b1 = load_frag16(wm + (t * 2 + 1) * 512 + lane * 16);
            acc0 = __builtin_amdgcn_wmma_f32_16x16x32_bf16(false, a, false, b0,
                                                           (short)0, acc0, false, false);
            acc1 = __builtin_amdgcn_wmma_f32_16x16x32_bf16(false, a, false, b1,
                                                           (short)0, acc1, false, false);
        }
    }

    __syncthreads();   // fn/fe region is about to be reused as tp/h scratch

    // ---------- 6) epilogue: tp -> fc1 -> fc2, all WMMA ----------
    // C/D layout: VGPR v, lane -> (M = v + 8*half, N = lane%16 [+16*nt]).
    float* tp = (float*)(smem + TP_OFF) + w * (16 * 32);
    const float s_tp = 1.0f / 45.0f;
#pragma unroll
    for (int v = 0; v < 8; ++v) {
        int m = v + half * 8;
        tp[m * 32 + lm]      = acc0[v] * s_tp;
        tp[m * 32 + 16 + lm] = acc1[v] * s_tp;
    }
    bf16x16 atp;
#pragma unroll
    for (int s = 0; s < 16; ++s) {
        int kk = s + (s & 8) + (half << 3);      // A layout: K = s + (s&8) + 8*half
        atp[s] = (__bf16)tp[lm * 32 + kk];
    }
    // h = tp @ W_fc1 / sqrt(32)   (K = 32 exactly -> 4 WMMAs)
    const __bf16* wf1 = (const __bf16*)(smem + WFC1_OFF);
    float* hbuf = (float*)(smem + H_OFF) + w * (16 * 64);
    const float s_h = 0.17677669529663687f;      // 1/sqrt(32)
#pragma unroll
    for (int nt = 0; nt < 4; ++nt) {
        f32x8 z = {};
        bf16x16 b = load_frag16(wf1 + nt * 512 + lane * 16);
        f32x8 hacc = __builtin_amdgcn_wmma_f32_16x16x32_bf16(false, atp, false, b,
                                                             (short)0, z, false, false);
#pragma unroll
        for (int v = 0; v < 8; ++v) {
            int m = v + half * 8;
            hbuf[m * 64 + nt * 16 + lm] = hacc[v] * s_h;
        }
    }
    // out = h @ W_fc2 / 8   (K = 64 -> 2 ksteps, N = 48 -> 3 ntiles, 6 WMMAs)
    const __bf16* wf2 = (const __bf16*)(smem + WFC2_OFF);
    f32x8 oacc[3] = {{}, {}, {}};
#pragma unroll
    for (int kt = 0; kt < 2; ++kt) {
        bf16x16 ah;
#pragma unroll
        for (int s = 0; s < 16; ++s) {
            int kk = kt * 32 + s + (s & 8) + (half << 3);
            ah[s] = (__bf16)hbuf[lm * 64 + kk];
        }
#pragma unroll
        for (int nt = 0; nt < 3; ++nt) {
            bf16x16 b = load_frag16(wf2 + (kt * 3 + nt) * 512 + lane * 16);
            oacc[nt] = __builtin_amdgcn_wmma_f32_16x16x32_bf16(false, ah, false, b,
                                                               (short)0, oacc[nt], false, false);
        }
    }
    // ---------- 7) store (guarded; WMMAs are done, divergence is fine) ----------
    const float s_o = 0.125f;                    // 1/sqrt(64)
    const long ecc0 = (long)blockIdx.x * 128 + w * 16;
#pragma unroll
    for (int nt = 0; nt < 3; ++nt) {
        int n = nt * 16 + lm;
        if (n < REP) {
#pragma unroll
            for (int v = 0; v < 8; ++v) {
                long ecc = ecc0 + v + half * 8;
                if (ecc < necc)
                    out[ecc * REP + n] = oacc[nt][v] * s_o;
            }
        }
    }
}

extern "C" void kernel_launch(void* const* d_in, const int* in_sizes, int n_in,
                              void* d_out, int out_size, void* d_ws, size_t ws_size,
                              hipStream_t stream)
{
    const float* f_nodes    = (const float*)d_in[0];
    const float* f_edges    = (const float*)d_in[1];
    const int*   edge_index = (const int*)d_in[2];
    const float* W_tp       = (const float*)d_in[3];
    const float* W_fc1      = (const float*)d_in[4];
    const float* W_fc2      = (const float*)d_in[5];
    float* out = (float*)d_out;

    const int n_edges = in_sizes[2] / 2;          // edge_index is (2, E)
    const long necc   = (long)n_edges * 2;
    const int grid    = (int)((necc + 127) / 128);

    (void)hipFuncSetAttribute((const void*)node_equi_tp_wmma,
                              hipFuncAttributeMaxDynamicSharedMemorySize,
                              SMEM_BYTES);

    hipLaunchKernelGGL(node_equi_tp_wmma, dim3(grid), dim3(256), SMEM_BYTES, stream,
                       f_nodes, f_edges, edge_index, W_tp, W_fc1, W_fc2, out, n_edges);
}
